// SingleHeadAttention_89816356094016
// MI455X (gfx1250) — matrix operand order repaired and checked by reading, not compiled, not run on hardware
//
#include <hip/hip_runtime.h>
#include <hip/hip_bf16.h>

// ---------------------------------------------------------------------------
// Single-head causal attention for MI455X (gfx1250), wave32 + WMMA bf16 + TDM.
//   x:[4,4096,2048] f32, W*:[2048,128] f32, b*:[128] f32 -> out:[4,4096,128] f32
// Pipeline:
//   1) wt_kernel   : W -> Wt bf16 [3][128][2048] (N-major for B-fragments)
//   2) proj_kernel : Q,K bf16 [B][T][128] row-major ; V bf16 transposed [B][128][T]
//   3) attn_kernel : flash attention; K/V tiles staged to LDS by the Tensor
//      Data Mover (tensor_load_to_lds, double-buffered, s_wait_tensorcnt),
//      QK^T and PV via v_wmma_f32_16x16x32_bf16.
// ---------------------------------------------------------------------------

typedef __attribute__((ext_vector_type(16))) __bf16 v16bf;
typedef __attribute__((ext_vector_type(8)))  float  v8f;
typedef __attribute__((ext_vector_type(4)))  unsigned int u32x4;
typedef __attribute__((ext_vector_type(8)))  int i32x8;
typedef __attribute__((ext_vector_type(4)))  int i32x4;

#define D_MODEL 2048
#define HEAD    128
#define SEQ     4096
#define BATCH   4

#if defined(__has_builtin)
#if __has_builtin(__builtin_amdgcn_tensor_load_to_lds)
#define USE_TDM 1
#endif
#endif

union ABFrag {
    v16bf v;
    uint4 u[2];
};

static __device__ inline v8f wmma_bf16(v16bf a, v16bf b, v8f c) {
    // D = A(16x32) * B(32x16) + C, f32 accumulate
    return __builtin_amdgcn_wmma_f32_16x16x32_bf16(
        /*neg_a=*/false, a, /*neg_b=*/false, b,
        /*c_mod=*/(short)0, c, /*reuse_a=*/false, /*reuse_b=*/false);
}

static __device__ inline v8f vzero8() {
    v8f z = {0.f, 0.f, 0.f, 0.f, 0.f, 0.f, 0.f, 0.f};
    return z;
}

#ifdef USE_TDM
// 2-D TDM load: tile (tl1 rows x tl0 elems) of a bf16 tensor whose innermost
// dim is td0 elems with row stride str0 elems, into LDS at lds_off (packed,
// row-major, tl0 elems per row). D# layout per CDNA5 ISA 8.3/8.4.
// Toolchain builtin is the 6-arg form:
//   (u32x4 group0, i32x8 group1, i32x4 group2, i32x4 group3, i32x8, i32 cpol)
static __device__ __attribute__((always_inline))
void tdm_load_2d(unsigned lds_off, const void* gptr,
                 unsigned td0, unsigned td1,
                 unsigned tl0, unsigned tl1, unsigned str0) {
    unsigned long long ga = (unsigned long long)gptr;
    u32x4 g0;
    g0[0] = 1u;                                    // count=1, user mode
    g0[1] = lds_off;                               // lds_addr (bytes)
    g0[2] = (unsigned)ga;                          // global_addr[31:0]
    g0[3] = ((unsigned)(ga >> 32) & 0x01FFFFFFu)   // global_addr[56:32]
            | 0x80000000u;                         // type=2 ("image")
    i32x8 g1;
    g1[0] = (int)(1u << 16);                       // data_size=1 (2 bytes)
    g1[1] = (int)((td0 & 0xFFFFu) << 16);          // tensor_dim0[15:0]
    g1[2] = (int)((td0 >> 16) | ((td1 & 0xFFFFu) << 16));
    g1[3] = (int)((td1 >> 16) | (tl0 << 16));      // tile_dim0
    g1[4] = (int)tl1;                              // tile_dim1 (tile_dim2=0)
    g1[5] = (int)str0;                             // tensor_dim0_stride[31:0]
    g1[6] = 0;
    g1[7] = 0;
    i32x4 z;
    z[0] = 0; z[1] = 0; z[2] = 0; z[3] = 0;        // groups 2/3 unused (2-D)
    i32x8 z8;
    z8[0] = 0; z8[1] = 0; z8[2] = 0; z8[3] = 0;
    z8[4] = 0; z8[5] = 0; z8[6] = 0; z8[7] = 0;
    __builtin_amdgcn_tensor_load_to_lds(g0, g1, z, z, z8, 0);
}

static __device__ __attribute__((always_inline))
unsigned lds_offset(const void* p) {
    return (unsigned)(unsigned long long)p;        // low 32 bits = LDS offset
}
#endif

// ---------------------------------------------------------------------------
// Kernel 1: convert+transpose the three weight matrices to bf16, N-major.
// Wt[sel][h][d] = W_sel[d][h]
// ---------------------------------------------------------------------------
__global__ __launch_bounds__(256)
void wt_kernel(const float* __restrict__ Wq, const float* __restrict__ Wk,
               const float* __restrict__ Wv, __bf16* __restrict__ Wt) {
    int idx = blockIdx.x * 256 + threadIdx.x;          // [0, 3*128*2048)
    int sel = idx / (HEAD * D_MODEL);
    int r   = idx - sel * (HEAD * D_MODEL);
    int h   = r / D_MODEL;
    int d   = r - h * D_MODEL;
    const float* W = (sel == 0) ? Wq : ((sel == 1) ? Wk : Wv);
    Wt[idx] = (__bf16)W[d * HEAD + h];
}

// ---------------------------------------------------------------------------
// Kernel 2: projection GEMM. Block = 64 rows x 128 cols for one of {Q,K,V}.
// 8 waves arranged 4(M) x 2(N); each wave owns 16 rows x 64 cols (4 WMMA tiles).
// ---------------------------------------------------------------------------
__global__ __launch_bounds__(256)
void proj_kernel(const float* __restrict__ x, const __bf16* __restrict__ Wt,
                 const float* __restrict__ bq, const float* __restrict__ bk,
                 const float* __restrict__ bv,
                 __bf16* __restrict__ Qb, __bf16* __restrict__ Kb,
                 __bf16* __restrict__ Vt) {
    const int sel = blockIdx.y;          // 0=Q 1=K 2=V
    const int b   = blockIdx.z;
    const int tid = threadIdx.x;
    const int wid = tid >> 5;
    const int lid = tid & 31;
    const int ln  = lid & 15;            // N index (B/C layout), row-in-tile (A layout)
    const int hi  = lid >> 4;            // lane half
    const int wm  = wid & 3;
    const int wn  = wid >> 2;

    const int m0   = blockIdx.x * 64 + wm * 16;
    const int n0   = wn * 64;
    const int arow = m0 + ln;
    const int kh_a = hi * 8;             // A-frag k-half offset
    const int kh_b = hi * 16;            // unused (B-frag loads also use kh_a)

    const float*  xrow = x + ((size_t)b * SEQ + arow) * D_MODEL;
    const __bf16* Wsel = Wt + (size_t)sel * HEAD * D_MODEL;

    v8f acc[4];
#pragma unroll
    for (int nt = 0; nt < 4; ++nt) acc[nt] = vzero8();

    for (int kb = 0; kb < D_MODEL; kb += 32) {
        // ---- batch all loads for this k-step ----
        float fl[16];
        const float4* ap  = (const float4*)(xrow + kb + kh_a);
        const float4* ap2 = (const float4*)(xrow + kb + kh_a + 16);
        float4 f0 = ap[0], f1 = ap[1], f2 = ap2[0], f3 = ap2[1];

        ABFrag bb[4];
#pragma unroll
        for (int nt = 0; nt < 4; ++nt) {
            const int wrow = n0 + nt * 16 + ln;
            const uint4* bp =
                (const uint4*)(Wsel + (size_t)wrow * D_MODEL + kb + kh_a);
            bb[nt].u[0] = bp[0];
            bb[nt].u[1] = bp[2];
        }

        *(float4*)(fl + 0)  = f0;
        *(float4*)(fl + 4)  = f1;
        *(float4*)(fl + 8)  = f2;
        *(float4*)(fl + 12) = f3;
        ABFrag a;
#pragma unroll
        for (int i = 0; i < 16; ++i) a.v[i] = (__bf16)fl[i];

        // ---- 4 back-to-back WMMAs ----
#pragma unroll
        for (int nt = 0; nt < 4; ++nt)
            acc[nt] = wmma_bf16(a.v, bb[nt].v, acc[nt]);
    }

    // ---- bias add + store ----
    const float* bias = (sel == 0) ? bq : ((sel == 1) ? bk : bv);
#pragma unroll
    for (int nt = 0; nt < 4; ++nt) {
        const int col = n0 + nt * 16 + ln;
        const float bz = bias[col];
        if (sel == 2) {
            // V stored transposed: Vt[b][col][row]; 8 consecutive rows per lane
            const int rbase = m0 + hi * 8;
            __bf16 vals[8];
#pragma unroll
            for (int j = 0; j < 8; ++j) vals[j] = (__bf16)(acc[nt][j] + bz);
            __bf16* dst = Vt + ((size_t)b * HEAD + col) * SEQ + rbase;
            *(uint4*)dst = *(const uint4*)vals;
        } else {
            __bf16* base = (sel == 0) ? Qb : Kb;
#pragma unroll
            for (int j = 0; j < 8; ++j) {
                const int row = m0 + j + hi * 8;
                base[((size_t)b * SEQ + row) * HEAD + col] =
                    (__bf16)(acc[nt][j] + bz);
            }
        }
    }
}

// ---------------------------------------------------------------------------
// Kernel 3: flash attention. Block = 64 query rows, 4 waves x 16 rows each,
// key blocks of 64 (diagonal block = last iteration). K/V tiles staged into
// double-buffered LDS by the Tensor Data Mover (wave 0 drives the DMA,
// s_wait_tensorcnt + workgroup barrier publish the tiles); fragments are then
// read with ds_load_b128. Per-wave LDS bounce converts P (C-layout) into
// A-fragment layout for the PV WMMAs (same-wave DS ordering).
// ---------------------------------------------------------------------------
__global__ __launch_bounds__(128)
void attn_kernel(const __bf16* __restrict__ Qb, const __bf16* __restrict__ Kb,
                 const __bf16* __restrict__ Vt, float* __restrict__ out) {
#ifdef USE_TDM
    __shared__ __align__(16) __bf16 KT[2][64 * HEAD];   // 2 x 16 KB
    __shared__ __align__(16) __bf16 VT[2][HEAD * 64];   // 2 x 16 KB
#endif
    __shared__ __align__(16) __bf16 Pld[4 * 16 * 64];   // 8 KB

    const int b   = blockIdx.y;
    const int qb  = blockIdx.x;                          // 64-row query block
    const int tid = threadIdx.x;
    const int wid = tid >> 5;                            // 0..3
    const int lid = tid & 31;
    const int ln  = lid & 15;
    const int hi  = lid >> 4;
    const int kh_a = hi * 8;
    const int kh_b = hi * 16;

    const int r0 = qb * 64 + wid * 16;    // first query row owned by this wave
    __bf16* myP  = Pld + wid * 16 * 64;

    // ---- load Q fragments for this wave's 16 rows (4 x 16x32 bf16) ----
    ABFrag qf[4];
    {
        const __bf16* qrow = Qb + ((size_t)b * SEQ + r0 + ln) * HEAD;
#pragma unroll
        for (int f = 0; f < 4; ++f) {
            const __bf16* p = qrow + f * 32 + kh_a;
            qf[f].u[0] = *(const uint4*)p;
            qf[f].u[1] = *(const uint4*)(p + 16);
        }
    }

    float mrow[8], lsum[8];
    v8f o[8];
#pragma unroll
    for (int j = 0; j < 8; ++j) { mrow[j] = -1e30f; lsum[j] = 0.f; }
#pragma unroll
    for (int vn = 0; vn < 8; ++vn) o[vn] = vzero8();

    const float scale = 0.08838834764831845f;   // 1/sqrt(128)

#ifdef USE_TDM
    const __bf16* kg = Kb + (size_t)b * SEQ * HEAD;   // [SEQ][HEAD]
    const __bf16* vg = Vt + (size_t)b * HEAD * SEQ;   // [HEAD][SEQ]
    if (wid == 0) {
        // prologue: DMA key block 0 into buffer 0
        tdm_load_2d(lds_offset(&KT[0][0]), kg,
                    HEAD, SEQ, HEAD, 64, HEAD);
        tdm_load_2d(lds_offset(&VT[0][0]), vg,
                    SEQ, HEAD, 64, HEAD, SEQ);
    }
#endif

    for (int kblk = 0; kblk <= qb; ++kblk) {
        const int key0 = kblk * 64;
        const int cur  = kblk & 1;

#ifdef USE_TDM
        if (wid == 0) {
            if (kblk < qb) {
                const int nk = key0 + 64;
                tdm_load_2d(lds_offset(&KT[cur ^ 1][0]),
                            kg + (size_t)nk * HEAD, HEAD, SEQ, HEAD, 64, HEAD);
                tdm_load_2d(lds_offset(&VT[cur ^ 1][0]),
                            vg + nk, SEQ, HEAD, 64, HEAD, SEQ);
                __builtin_amdgcn_s_wait_tensorcnt(2);  // current block landed
            } else {
                __builtin_amdgcn_s_wait_tensorcnt(0);
            }
        }
        __syncthreads();                                // publish LDS tiles
        const __bf16* kbase = &KT[cur][0];  const int kstr = HEAD;
        const __bf16* vbase = &VT[cur][0];  const int vstr = 64;
#else
        const __bf16* kbase = Kb + ((size_t)b * SEQ + key0) * HEAD;
        const int kstr = HEAD;
        const __bf16* vbase = Vt + (size_t)b * HEAD * SEQ + key0;
        const int vstr = SEQ;
        if (kblk < qb) {
            __builtin_prefetch(kbase + (size_t)64 * HEAD + kh_b, 0, 3);
        }
#endif

        // ---- S = Q * K^T : per N-tile, batch-load 4 K frags then 4 WMMAs ----
        v8f s[4];
#pragma unroll
        for (int nt = 0; nt < 4; ++nt) {
            const __bf16* krow = kbase + (size_t)(nt * 16 + ln) * kstr;
            ABFrag kf[4];
#pragma unroll
            for (int f = 0; f < 4; ++f) {
                const __bf16* p = krow + f * 32 + kh_a;
                kf[f].u[0] = *(const uint4*)p;
                kf[f].u[1] = *(const uint4*)(p + 16);
            }
            v8f acc = vzero8();
#pragma unroll
            for (int f = 0; f < 4; ++f)
                acc = wmma_bf16(qf[f].v, kf[f].v, acc);
            s[nt] = acc;
        }

        // ---- scale + causal mask (only the diagonal block needs masking) ----
        const bool diag = (kblk == qb);
#pragma unroll
        for (int nt = 0; nt < 4; ++nt) {
#pragma unroll
            for (int j = 0; j < 8; ++j) {
                float v = s[nt][j] * scale;
                if (diag) {
                    const int col = key0 + nt * 16 + ln;
                    const int row = r0 + j + hi * 8;
                    if (col > row) v = -1e30f;
                }
                s[nt][j] = v;
            }
        }

        // ---- online softmax: row max (in-lane over nt, then 16-lane halves) ----
        float mnew[8];
#pragma unroll
        for (int j = 0; j < 8; ++j) {
            float mv = mrow[j];
#pragma unroll
            for (int nt = 0; nt < 4; ++nt) mv = fmaxf(mv, s[nt][j]);
            mv = fmaxf(mv, __shfl_xor(mv, 1));
            mv = fmaxf(mv, __shfl_xor(mv, 2));
            mv = fmaxf(mv, __shfl_xor(mv, 4));
            mv = fmaxf(mv, __shfl_xor(mv, 8));
            mnew[j] = mv;
        }

        // rescale running state
#pragma unroll
        for (int j = 0; j < 8; ++j) {
            const float corr = __expf(mrow[j] - mnew[j]);
            lsum[j] *= corr;
#pragma unroll
            for (int vn = 0; vn < 8; ++vn) o[vn][j] *= corr;
            mrow[j] = mnew[j];
        }

        // ---- P = exp(S - m), accumulate row sums, stage bf16 P to LDS ----
        float psum[8];
#pragma unroll
        for (int j = 0; j < 8; ++j) psum[j] = 0.f;
#pragma unroll
        for (int nt = 0; nt < 4; ++nt) {
#pragma unroll
            for (int j = 0; j < 8; ++j) {
                const float p = __expf(s[nt][j] - mrow[j]);
                psum[j] += p;
                myP[(j + hi * 8) * 64 + nt * 16 + ln] = (__bf16)p;
            }
        }
#pragma unroll
        for (int j = 0; j < 8; ++j) {
            float sv = psum[j];
            sv += __shfl_xor(sv, 1);
            sv += __shfl_xor(sv, 2);
            sv += __shfl_xor(sv, 4);
            sv += __shfl_xor(sv, 8);
            lsum[j] += sv;
        }

        // ---- reload P as A-fragments (row-major in LDS -> A layout) ----
        ABFrag pf[2];
        {
            const __bf16* prow = myP + ln * 64;
#pragma unroll
            for (int f = 0; f < 2; ++f) {
                const __bf16* p = prow + f * 32 + kh_a;
                pf[f].u[0] = *(const uint4*)p;
                pf[f].u[1] = *(const uint4*)(p + 16);
            }
        }

        // ---- O += P * V : per head tile, batch-load 2 V frags then 2 WMMAs ----
#pragma unroll
        for (int vn = 0; vn < 8; ++vn) {
            const __bf16* vrow = vbase + (size_t)(vn * 16 + ln) * vstr;
            ABFrag vf[2];
#pragma unroll
            for (int f = 0; f < 2; ++f) {
                const __bf16* p = vrow + f * 32 + kh_a;
                vf[f].u[0] = *(const uint4*)p;
                vf[f].u[1] = *(const uint4*)(p + 16);
            }
            v8f acc = o[vn];
#pragma unroll
            for (int f = 0; f < 2; ++f)
                acc = wmma_bf16(pf[f].v, vf[f].v, acc);
            o[vn] = acc;
        }

#ifdef USE_TDM
        __syncthreads();   // all waves done reading buf[cur] before it is
                           // re-targeted by the DMA in the next iteration
#endif
    }

    // ---- normalize and write out (f32) ----
#pragma unroll
    for (int j = 0; j < 8; ++j) {
        const float inv = 1.0f / lsum[j];
        const size_t rowoff = ((size_t)b * SEQ + r0 + j + hi * 8) * HEAD;
#pragma unroll
        for (int vn = 0; vn < 8; ++vn) {
            out[rowoff + vn * 16 + ln] = o[vn][j] * inv;
        }
    }
}

// ---------------------------------------------------------------------------
extern "C" void kernel_launch(void* const* d_in, const int* in_sizes, int n_in,
                              void* d_out, int out_size, void* d_ws, size_t ws_size,
                              hipStream_t stream) {
    (void)in_sizes; (void)n_in; (void)out_size; (void)ws_size;

    const float* x  = (const float*)d_in[0];
    const float* Wq = (const float*)d_in[1];
    const float* bq = (const float*)d_in[2];
    const float* Wk = (const float*)d_in[3];
    const float* bk = (const float*)d_in[4];
    const float* Wv = (const float*)d_in[5];
    const float* bv = (const float*)d_in[6];
    float* out = (float*)d_out;

    char* ws = (char*)d_ws;
    const size_t wt_bytes  = (size_t)3 * HEAD * D_MODEL * 2;       // 1.5 MB
    const size_t qkv_elems = (size_t)BATCH * SEQ * HEAD;           // 2 Mi elems
    __bf16* Wt = (__bf16*)ws;
    __bf16* Qb = (__bf16*)(ws + wt_bytes);
    __bf16* Kb = Qb + qkv_elems;
    __bf16* Vt = Kb + qkv_elems;                                    // [B][H][T]

    // 1) weight convert/transpose: 3*128*2048 elements
    wt_kernel<<<dim3((3 * HEAD * D_MODEL) / 256), dim3(256), 0, stream>>>(
        Wq, Wk, Wv, Wt);

    // 2) projections: (T/64) strips x 3 matrices x B batches
    proj_kernel<<<dim3(SEQ / 64, 3, BATCH), dim3(256), 0, stream>>>(
        x, Wt, bq, bk, bv, Qb, Kb, Vt);

    // 3) flash attention: (T/64) query blocks x B batches = 256 workgroups
    attn_kernel<<<dim3(SEQ / 64, BATCH), dim3(128), 0, stream>>>(
        Qb, Kb, Vt, out);
}
